// Model_36687610643120
// MI455X (gfx1250) — matrix operand order repaired and checked by reading, not compiled, not run on hardware
//
#include <hip/hip_runtime.h>

// ---------------- types / constants ----------------
typedef __attribute__((ext_vector_type(16))) _Float16 v16h;
typedef __attribute__((ext_vector_type(8)))  float    v8f;
typedef unsigned int u32x4 __attribute__((ext_vector_type(4)));
typedef int          i32x8 __attribute__((ext_vector_type(8)));
typedef int          i32x4 __attribute__((ext_vector_type(4)));

#define BATCH   64
#define SEQ     512
#define EMB     256
#define GATES   1024      // 4*H
#define NTAG    9
#define HPAD    264       // padded LDS row stride for h (f16)
#define XS_OFF  (BATCH * HPAD * 2)            // 33792 B: start of xp slabs
#define XS_BYTES (BATCH * GATES * 2)          // 131072 B per f16 slab

#if defined(__HIP_DEVICE_COMPILE__) && defined(__gfx1250__) && \
    __has_builtin(__builtin_amdgcn_tensor_load_to_lds)
#define HAS_TDM 1
#else
#define HAS_TDM 0
#endif

union Frag { uint4 q[2]; v16h h; };

__device__ __forceinline__ v8f wmma_f16(v16h a, v16h b, v8f c) {
  return __builtin_amdgcn_wmma_f32_16x16x32_f16(
      false, a, false, b, (short)0, c, false, false);
}

// A fragment: rows m0..m0+15 of a row-major f16 matrix, K-slice [k0,k0+32)
__device__ __forceinline__ v16h load_a(const _Float16* base, long rowStride,
                                       int k0, int lane) {
  int m = lane & 15, sel = lane >> 4;
  const _Float16* p = base + (long)m * rowStride + k0 + sel * 8;
  Frag f;
  f.q[0] = *(const uint4*)(p);
  f.q[1] = *(const uint4*)(p + 16);
  return f.h;
}

// B fragment from W stored row-major [N,K]
__device__ __forceinline__ v16h load_b(const _Float16* W, int ldw,
                                       int n0, int k0, int lane) {
  int n = lane & 15, sel = lane >> 4;
  const _Float16* p = W + (long)(n0 + n) * ldw + k0 + sel * 8;
  Frag f;
  f.q[0] = *(const uint4*)(p);
  f.q[1] = *(const uint4*)(p + 16);
  return f.h;
}

#if HAS_TDM
// Issue one TDM DMA of a contiguous slab (bytes = units*8) into LDS.
// D# group0: count=1 | lds_addr | global_addr | type=2("image")
// D# group1: data_size=3 (8B units); 1-row tile: tile_dim0=units, tile_dim1=1
__device__ __forceinline__ void tdm_load_slab(const void* gsrc,
                                              unsigned lds_off, unsigned units) {
  unsigned long long ga = (unsigned long long)gsrc;
  u32x4 g0;
  g0[0] = 1u;                                   // count=1, user mode
  g0[1] = lds_off;                              // LDS byte address
  g0[2] = (unsigned)ga;                         // global addr [31:0]
  g0[3] = (unsigned)(ga >> 32) | (2u << 30);    // addr [56:32] | type=2
  i32x8 g1;
  g1[0] = (int)(3u << 16);                      // data_size=3 (8 bytes)
  g1[1] = (int)((units & 0xFFFFu) << 16);       // tensor_dim0[15:0]
  g1[2] = (int)((units >> 16) | (1u << 16));    // tensor_dim0[31:16] | tensor_dim1=1
  g1[3] = (int)((units & 0xFFFFu) << 16);       // tile_dim0 = units (<=65535)
  g1[4] = 1;                                    // tile_dim1 = 1
  g1[5] = (int)units;                           // tensor_dim0_stride[31:0]
  g1[6] = 0;
  g1[7] = 0;
  i32x4 z4 = {0, 0, 0, 0};
  i32x8 z8 = {0, 0, 0, 0, 0, 0, 0, 0};
  // clang-23 / therock-10.0 signature: (g0, g1, g2, g3, g4, cpol)
  __builtin_amdgcn_tensor_load_to_lds(g0, g1, z4, z4, z8, 0);
}
#endif

// ---------------- kernel 1: embedding gather + f16 cast ----------------
__global__ __launch_bounds__(256)
void k_embed(const int* __restrict__ ids, const float* __restrict__ table,
             _Float16* __restrict__ out) {
  long idx = (long)blockIdx.x * 256 + threadIdx.x;
  int row = (int)(idx >> 8);
  int e   = (int)(idx & 255);
  int id  = ids[row];
  out[idx] = (_Float16)table[(long)id * EMB + e];
}

// ---------------- kernel 2: f32 -> f16 weight cast ----------------
__global__ __launch_bounds__(256)
void k_cast16(const float* __restrict__ src, _Float16* __restrict__ dst, int n) {
  int i = blockIdx.x * 256 + threadIdx.x;
  if (i < n) dst[i] = (_Float16)src[i];
}

// ---------------- kernel 3: input projection GEMM (WMMA), f16 output ----
// xp[dir][s][b][n] = emb[b][t(s,dir)][:] . w_ih[dir][n][:] + bias[dir][n]
__global__ __launch_bounds__(256)
void k_proj(const _Float16* __restrict__ emb,
            const _Float16* __restrict__ wih_f, const float* __restrict__ bf,
            const _Float16* __restrict__ wih_b, const float* __restrict__ bb,
            _Float16* __restrict__ xp_f, _Float16* __restrict__ xp_b) {
  int lane = threadIdx.x & 31;
  long gw  = (long)blockIdx.x * 8 + (threadIdx.x >> 5);
  int  mt  = (int)(gw & 2047);
  long rs  = gw >> 11;
  int  nt  = (int)(rs & 63);
  int  dir = (int)(rs >> 6);

  int s  = mt >> 2;
  int b0 = (mt & 3) * 16;
  int ta = dir ? (SEQ - 1 - s) : s;
  int n0 = nt * 16;

  const _Float16* A    = emb + ((long)b0 * SEQ + ta) * EMB;
  const _Float16* W    = dir ? wih_b : wih_f;
  const float*    bias = dir ? bb : bf;
  _Float16*       out  = (dir ? xp_b : xp_f) + (long)s * BATCH * GATES;

  int col = lane & 15, rb = (lane >> 4) * 8;
  float bv = bias[n0 + col];
  v8f c;
#pragma unroll
  for (int r = 0; r < 8; ++r) c[r] = bv;

#pragma unroll
  for (int kk = 0; kk < 8; ++kk) {
    int k0 = kk * 32;
    v16h a   = load_a(A, (long)SEQ * EMB, k0, lane);
    v16h bfr = load_b(W, EMB, n0, k0, lane);
    c = wmma_f16(a, bfr, c);
  }
#pragma unroll
  for (int r = 0; r < 8; ++r)
    out[(long)(b0 + rb + r) * GATES + n0 + col] = (_Float16)c[r];
}

// ---------------- kernel 4: persistent LSTM scan --------------------------
// 1 workgroup (32 waves) per direction. LDS: h[64][HPAD] f16 (33 KB) +
// two TDM-filled xp slabs (2x128 KB). Each wave owns 2 groups of 4 gate
// tiles (i,f,g,o) for the same (m-tile, j-tile); the LSTM cell runs on the
// WMMA accumulators in registers and c never leaves VGPRs.
__global__ __launch_bounds__(1024)
void k_scan(const _Float16* __restrict__ xp_f, const _Float16* __restrict__ xp_b,
            const _Float16* __restrict__ whh_f, const _Float16* __restrict__ whh_b,
            _Float16* __restrict__ hs_f, _Float16* __restrict__ hs_b) {
  extern __shared__ char smem[];
  _Float16* hsh = (_Float16*)smem;                   // [64][HPAD]

  int dir = blockIdx.x;
  const _Float16* xp = dir ? xp_b : xp_f;
  const _Float16* W  = dir ? whh_b : whh_f;
  _Float16*       hs = dir ? hs_b : hs_f;

  int tid = threadIdx.x, lane = tid & 31, w = tid >> 5;
  int col = lane & 15, rb = (lane >> 4) * 8;
  int m0  = (w & 3) * 16;        // batch tile
  int j0a = (w >> 2) * 16;       // hidden-column tiles (two per wave)
  int j0b = j0a + 128;

  for (int i = tid; i < BATCH * HPAD; i += 1024) hsh[i] = (_Float16)0.f;

  v8f creg[2];
#pragma unroll
  for (int g = 0; g < 2; ++g)
#pragma unroll
    for (int r = 0; r < 8; ++r) creg[g][r] = 0.f;

#if HAS_TDM
  if (w == 0) tdm_load_slab(xp, XS_OFF, XS_BYTES / 8);   // preload slab 0
#endif
  __syncthreads();

  for (int t = 0; t < SEQ; ++t) {
    int stage = 0;
#if HAS_TDM
    stage = t & 1;
    if (w == 0) __builtin_amdgcn_s_wait_tensorcnt(0);    // slab t landed
    __syncthreads();                                     // publish + t-1 reads done
    if (w == 0 && t + 1 < SEQ)                           // overlap DMA of t+1
      tdm_load_slab(xp + (long)(t + 1) * BATCH * GATES,
                    XS_OFF + (unsigned)((t + 1) & 1) * XS_BYTES, XS_BYTES / 8);
#else
    {
      const uint4* src = (const uint4*)(xp + (long)t * BATCH * GATES);
      uint4* dst = (uint4*)(smem + XS_OFF);
      for (int i = tid; i < XS_BYTES / 16; i += 1024) dst[i] = src[i];
      __syncthreads();
    }
#endif
    const _Float16* xs = (const _Float16*)(smem + XS_OFF + stage * XS_BYTES);

    // ---- gates = xp[t] + h @ w_hh^T : 8 WMMA tiles per wave, accs in regs
    v8f acc[2][4];
#pragma unroll
    for (int g = 0; g < 2; ++g) {
      int j0 = g ? j0b : j0a;
#pragma unroll
      for (int q = 0; q < 4; ++q)
#pragma unroll
        for (int r = 0; r < 8; ++r)
          acc[g][q][r] = (float)xs[(m0 + rb + r) * GATES + j0 + q * 256 + col];
    }
#pragma unroll
    for (int kk = 0; kk < 8; ++kk) {
      int k0 = kk * 32;
      v16h a = load_a(hsh + (long)m0 * HPAD, HPAD, k0, lane);  // shared by 8 WMMAs
#pragma unroll
      for (int g = 0; g < 2; ++g) {
        int j0 = g ? j0b : j0a;
#pragma unroll
        for (int q = 0; q < 4; ++q) {
          v16h bfr = load_b(W, EMB, j0 + q * 256, k0, lane);   // L2-resident
          acc[g][q] = wmma_f16(a, bfr, acc[g][q]);
        }
      }
    }
    __syncthreads();   // all reads of h(t-1) and xs(t) complete

    // ---- fused LSTM cell on accumulators (torch gate order i,f,g,o)
#pragma unroll
    for (int g = 0; g < 2; ++g) {
      int j0 = g ? j0b : j0a;
#pragma unroll
      for (int r = 0; r < 8; ++r) {
        float gi = acc[g][0][r], gf = acc[g][1][r];
        float gg = acc[g][2][r], go = acc[g][3][r];
        float ii = 1.f / (1.f + __expf(-gi));
        float ff = 1.f / (1.f + __expf(-gf));
        float gv = tanhf(gg);
        float oo = 1.f / (1.f + __expf(-go));
        float cc = ff * creg[g][r] + ii * gv;
        creg[g][r] = cc;
        float hh = oo * tanhf(cc);
        int m = m0 + rb + r;
        hsh[m * HPAD + j0 + col] = (_Float16)hh;
        hs[((long)t * BATCH + m) * EMB + j0 + col] = (_Float16)hh;
      }
    }
    // next-iteration top barrier publishes h(t) before anyone reads it
#if !HAS_TDM
    __syncthreads();
#endif
  }
#if HAS_TDM
  __syncthreads();
#endif
}

// ---------------- kernel 5: LayerNorm + classifier + argmax ----------------
__global__ __launch_bounds__(128)
void k_lncls(const _Float16* __restrict__ hs_f, const _Float16* __restrict__ hs_b,
             const float* __restrict__ gamma, const float* __restrict__ beta,
             const float* __restrict__ cw, const float* __restrict__ cb,
             float* __restrict__ logits, float* __restrict__ preds) {
  int bt = blockIdx.x;            // b*SEQ + t
  int b = bt >> 9, t = bt & 511;
  __shared__ float xs[512];
  __shared__ float red[128];
  __shared__ float lg[16];
  int tid = threadIdx.x;

  for (int d = tid; d < 512; d += 128) {
    float v = (d < 256)
      ? (float)hs_f[((long)t * BATCH + b) * EMB + d]
      : (float)hs_b[((long)(SEQ - 1 - t) * BATCH + b) * EMB + (d - 256)];
    xs[d] = v;
  }
  __syncthreads();

  float s = 0.f;
  for (int d = tid; d < 512; d += 128) s += xs[d];
  red[tid] = s; __syncthreads();
  for (int o = 64; o; o >>= 1) { if (tid < o) red[tid] += red[tid + o]; __syncthreads(); }
  float mu = red[0] / 512.f;
  __syncthreads();

  float v2 = 0.f;
  for (int d = tid; d < 512; d += 128) { float dd = xs[d] - mu; v2 += dd * dd; }
  red[tid] = v2; __syncthreads();
  for (int o = 64; o; o >>= 1) { if (tid < o) red[tid] += red[tid + o]; __syncthreads(); }
  float rstd = rsqrtf(red[0] / 512.f + 1e-5f);
  __syncthreads();

  for (int d = tid; d < 512; d += 128)
    xs[d] = (xs[d] - mu) * rstd * gamma[d] + beta[d];
  __syncthreads();

  float acc[NTAG];
#pragma unroll
  for (int k = 0; k < NTAG; ++k) acc[k] = 0.f;
  for (int d = tid; d < 512; d += 128) {
    float x = xs[d];
#pragma unroll
    for (int k = 0; k < NTAG; ++k) acc[k] += x * cw[k * 512 + d];
  }
  for (int k = 0; k < NTAG; ++k) {
    red[tid] = acc[k]; __syncthreads();
    for (int o = 64; o; o >>= 1) { if (tid < o) red[tid] += red[tid + o]; __syncthreads(); }
    if (tid == 0) lg[k] = red[0] + cb[k];
    __syncthreads();
  }
  if (tid == 0) {
    float best = lg[0]; int bi = 0;
#pragma unroll
    for (int k = 0; k < NTAG; ++k) {
      logits[(long)bt * NTAG + k] = lg[k];
      if (lg[k] > best) { best = lg[k]; bi = k; }
    }
    preds[bt] = (float)bi;
  }
}

// ---------------- kernel 6: CRF forward DP + gold score + loss -------------
__global__ __launch_bounds__(64)
void k_crf(const float* __restrict__ logits, const int* __restrict__ labels,
           const float* __restrict__ start, const float* __restrict__ endv,
           const float* __restrict__ trans, float* __restrict__ out_loss) {
  __shared__ float tr[81];
  __shared__ float st[NTAG], en[NTAG];
  __shared__ float red[64];
  int tid = threadIdx.x;
  if (tid < 81) tr[tid] = trans[tid];
  if (tid < NTAG) { st[tid] = start[tid]; en[tid] = endv[tid]; }
  __syncthreads();

  int b = tid;
  const float* lg = logits + (long)b * SEQ * NTAG;
  const int*   lb = labels + (long)b * SEQ;

  float num = st[lb[0]] + en[lb[SEQ - 1]];
  for (int t = 0; t < SEQ; ++t)     num += lg[t * NTAG + lb[t]];
  for (int t = 0; t < SEQ - 1; ++t) num += tr[lb[t] * NTAG + lb[t + 1]];

  float alpha[NTAG];
#pragma unroll
  for (int k = 0; k < NTAG; ++k) alpha[k] = st[k] + lg[k];
  for (int t = 1; t < SEQ; ++t) {
    float na[NTAG];
#pragma unroll
    for (int j = 0; j < NTAG; ++j) {
      float mx = -1e30f;
#pragma unroll
      for (int i = 0; i < NTAG; ++i) mx = fmaxf(mx, alpha[i] + tr[i * NTAG + j]);
      float ssum = 0.f;
#pragma unroll
      for (int i = 0; i < NTAG; ++i) ssum += __expf(alpha[i] + tr[i * NTAG + j] - mx);
      na[j] = mx + __logf(ssum) + lg[t * NTAG + j];
    }
#pragma unroll
    for (int j = 0; j < NTAG; ++j) alpha[j] = na[j];
  }
  float mx = -1e30f;
#pragma unroll
  for (int k = 0; k < NTAG; ++k) { alpha[k] += en[k]; mx = fmaxf(mx, alpha[k]); }
  float ssum = 0.f;
#pragma unroll
  for (int k = 0; k < NTAG; ++k) ssum += __expf(alpha[k] - mx);
  float logZ = mx + __logf(ssum);

  red[tid] = num - logZ;
  __syncthreads();
  for (int o = 32; o; o >>= 1) { if (tid < o) red[tid] += red[tid + o]; __syncthreads(); }
  if (tid == 0) out_loss[0] = -red[0];
}

// ---------------- host launcher ----------------
extern "C" void kernel_launch(void* const* d_in, const int* in_sizes, int n_in,
                              void* d_out, int out_size, void* d_ws, size_t ws_size,
                              hipStream_t stream) {
  const int*   word_ids  = (const int*)d_in[0];
  const int*   label_ids = (const int*)d_in[1];
  const float* table     = (const float*)d_in[2];
  const float* w_ih_f    = (const float*)d_in[3];
  const float* w_hh_f    = (const float*)d_in[4];
  const float* b_f       = (const float*)d_in[5];
  const float* w_ih_b    = (const float*)d_in[6];
  const float* w_hh_b    = (const float*)d_in[7];
  const float* b_b       = (const float*)d_in[8];
  const float* ln_gamma  = (const float*)d_in[9];
  const float* ln_beta   = (const float*)d_in[10];
  const float* cls_w     = (const float*)d_in[11];
  const float* cls_b     = (const float*)d_in[12];
  const float* crf_start = (const float*)d_in[13];
  const float* crf_end   = (const float*)d_in[14];
  const float* crf_trans = (const float*)d_in[15];
  float* out = (float*)d_out;

  char* ws = (char*)d_ws;
  size_t off = 0;
  auto alloc = [&](size_t bytes) -> void* {
    void* p = ws + off;
    off = (off + bytes + 255) & ~(size_t)255;
    return p;
  };
  _Float16* emb16   = (_Float16*)alloc((size_t)BATCH * SEQ * EMB * 2);
  _Float16* wih_f16 = (_Float16*)alloc((size_t)GATES * EMB * 2);
  _Float16* whh_f16 = (_Float16*)alloc((size_t)GATES * EMB * 2);
  _Float16* wih_b16 = (_Float16*)alloc((size_t)GATES * EMB * 2);
  _Float16* whh_b16 = (_Float16*)alloc((size_t)GATES * EMB * 2);
  _Float16* xp_f    = (_Float16*)alloc((size_t)SEQ * BATCH * GATES * 2);
  _Float16* xp_b    = (_Float16*)alloc((size_t)SEQ * BATCH * GATES * 2);
  _Float16* hs_f    = (_Float16*)alloc((size_t)SEQ * BATCH * EMB * 2);
  _Float16* hs_b    = (_Float16*)alloc((size_t)SEQ * BATCH * EMB * 2);
  float*    logits  = (float*)alloc((size_t)BATCH * SEQ * NTAG * 4);

  k_embed<<<(BATCH * SEQ * EMB) / 256, 256, 0, stream>>>(word_ids, table, emb16);

  const int wn = GATES * EMB;
  k_cast16<<<(wn + 255) / 256, 256, 0, stream>>>(w_ih_f, wih_f16, wn);
  k_cast16<<<(wn + 255) / 256, 256, 0, stream>>>(w_hh_f, whh_f16, wn);
  k_cast16<<<(wn + 255) / 256, 256, 0, stream>>>(w_ih_b, wih_b16, wn);
  k_cast16<<<(wn + 255) / 256, 256, 0, stream>>>(w_hh_b, whh_b16, wn);

  k_proj<<<32768, 256, 0, stream>>>(emb16, wih_f16, b_f, wih_b16, b_b, xp_f, xp_b);

  // h (33 KB) + double-buffered TDM xp slabs (2 x 128 KB) = ~289 KB of WGP LDS
  const int smem = XS_OFF + 2 * XS_BYTES;
  k_scan<<<2, 1024, smem, stream>>>(xp_f, xp_b, whh_f16, whh_b16, hs_f, hs_b);

  k_lncls<<<BATCH * SEQ, 128, 0, stream>>>(hs_f, hs_b, ln_gamma, ln_beta,
                                           cls_w, cls_b, logits, out + 1);

  k_crf<<<1, 64, 0, stream>>>(logits, label_ids, crf_start, crf_end, crf_trans, out);
}
